// HONAugmentor_73950746902810
// MI455X (gfx1250) — matrix operand order repaired and checked by reading, not compiled, not run on hardware
//
#include <hip/hip_runtime.h>

#define NN      10000
#define INDIM   128
#define HIDD    256
#define NEDGE   160000
#define KSEL    (2 * NEDGE)

typedef __attribute__((ext_vector_type(16))) __bf16 v16bf;
typedef __attribute__((ext_vector_type(8)))  __bf16 v8bf;
typedef __attribute__((ext_vector_type(8)))  float  v8f;

// ---------- helpers ----------
__device__ __forceinline__ __bf16 f2bf(float f) {
  unsigned u = __float_as_uint(f);
  u += 0x7FFFu + ((u >> 16) & 1u);            // round to nearest even
  unsigned short h = (unsigned short)(u >> 16);
  return __builtin_bit_cast(__bf16, h);
}
// order-preserving fp32 <-> uint32 (unsigned compare == float compare)
__device__ __forceinline__ unsigned ordf(float f) {
  unsigned b = __float_as_uint(f);
  return b ^ ((b & 0x80000000u) ? 0xFFFFFFFFu : 0x80000000u);
}
__device__ __forceinline__ float unordf(unsigned u) {
  unsigned b = (u & 0x80000000u) ? (u ^ 0x80000000u) : ~u;
  return __uint_as_float(b);
}

// ---------- generic utility kernels ----------
__global__ void k_fill_u32(unsigned* __restrict__ p, unsigned v, long long n) {
  long long i = (long long)blockIdx.x * blockDim.x + threadIdx.x;
  if (i < n) p[i] = v;
}
__global__ void k_copy_f32(const float* __restrict__ in, float* __restrict__ out, long long n) {
  long long i = (long long)blockIdx.x * blockDim.x + threadIdx.x;
  if (i < n) out[i] = in[i];
}

// ---------- GIN: edge aggregation (segment_sum of h[src] into dst) ----------
__global__ void k_agg(const float* __restrict__ h, const int* __restrict__ src,
                      const int* __restrict__ dst, float* __restrict__ agg,
                      int dim, int nE) {
  long long i = (long long)blockIdx.x * blockDim.x + threadIdx.x;
  const int chunks = dim >> 2;
  if (i >= (long long)nE * chunks) return;
  const int e = (int)(i / chunks);
  const int c = ((int)(i % chunks)) << 2;
  const int s = src[e], d = dst[e];
  const float4 v = *(const float4*)(h + (long long)s * dim + c);
  float* p = agg + (long long)d * dim + c;
  atomicAdd(p + 0, v.x); atomicAdd(p + 1, v.y);
  atomicAdd(p + 2, v.z); atomicAdd(p + 3, v.w);
}

// z = h + agg, converted to bf16 (A operand of next GEMM)
__global__ void k_add_cvt(const float* __restrict__ h, const float* __restrict__ agg,
                          __bf16* __restrict__ zb, long long n) {
  long long i = (long long)blockIdx.x * blockDim.x + threadIdx.x;
  if (i < n) zb[i] = f2bf(h[i] + agg[i]);
}

// Wt[n*Kd + k] = bf16(W[k*Nc + n])   (pre-transpose weights -> contiguous-K B operand)
__global__ void k_cvt_wt(const float* __restrict__ W, __bf16* __restrict__ Wt,
                         int Kd, int Nc) {
  int i = blockIdx.x * blockDim.x + threadIdx.x;
  if (i >= Kd * Nc) return;
  int n = i / Kd, k = i % Kd;
  Wt[i] = f2bf(W[(long long)k * Nc + n]);
}

// ---------- WMMA GEMM (fully specialized at compile time; no branches in K-loop) ----------
// out[M, Ncols] = A[M, KD](bf16 row-major) x Bt[Ncols, KD]^T(bf16, row n = column n, contiguous K).
// One wave computes a 16 x (16*NSUB) strip: NSUB f32 accumulators reuse one A fragment.
// Requirements: M % 16 == 0, KD % 32 == 0, strip fully inside Ncols (tail handled by NSUB=1 launch).
template <int KD, int NSUB, bool RELU, bool BIAS, bool OUTF, bool OUTB, bool GMAX>
__global__ void k_gemm_t(const __bf16* __restrict__ A, const __bf16* __restrict__ Bt,
                         const float* __restrict__ bias,
                         float* __restrict__ outF, __bf16* __restrict__ outB,
                         unsigned* __restrict__ gmax,
                         int M, int Ncols, int n0base) {
  const int lane = threadIdx.x & 31;
  const int wave = threadIdx.x >> 5;
  const int m0 = (blockIdx.x * 8 + wave) << 4;
  if (m0 >= M) return;                               // uniform per wave
  const int n0 = n0base + blockIdx.y * (16 * NSUB);
  const int half = lane >> 4;                        // 0: lanes 0-15, 1: lanes 16-31
  const int l16  = lane & 15;

  // A fragment base: lanes 0-15 -> K {k0..k0+15}; lanes 16-31 -> {k0+16..k0+31}
  const __bf16* __restrict__ aRow = A + (long long)(m0 + l16) * KD + (half << 4);
  // B fragment base per subtile: column = n0 + t*16 + l16, 16 contiguous K per half-wave
  const __bf16* bRow[NSUB];
#pragma unroll
  for (int t = 0; t < NSUB; ++t)
    bRow[t] = Bt + (long long)(n0 + (t << 4) + l16) * KD + (half << 4);

  v8f c[NSUB];
#pragma unroll
  for (int t = 0; t < NSUB; ++t) c[t] = v8f{};

#pragma unroll
  for (int k0 = 0; k0 < KD; k0 += 32) {
    v16bf a;
    *(v8bf*)&a       = *(const v8bf*)(aRow + k0);
    *((v8bf*)&a + 1) = *(const v8bf*)(aRow + k0 + 8);
#pragma unroll
    for (int t = 0; t < NSUB; ++t) {
      v16bf b = *(const v16bf*)(bRow[t] + k0);
      c[t] = __builtin_amdgcn_wmma_f32_16x16x32_bf16(
          false, a, false, b, (short)0, c[t], false, false);
    }
  }

  // epilogue: C layout -> element (m0 + half*8 + r, n0 + t*16 + l16) = c[t][r]
  float mx = -3.0e38f;
#pragma unroll
  for (int t = 0; t < NSUB; ++t) {
    const int ncol = n0 + (t << 4) + l16;
    float bv = 0.0f;
    if (BIAS) bv = bias[ncol];
#pragma unroll
    for (int r = 0; r < 8; ++r) {
      float v = c[t][r] + bv;
      if (RELU) v = fmaxf(v, 0.0f);
      const long long o = (long long)(m0 + (half << 3) + r) * Ncols + ncol;
      if (OUTF) outF[o] = v;
      if (OUTB) outB[o] = f2bf(v);
      if (GMAX) mx = fmaxf(mx, v);
    }
  }
  if (GMAX) {                                        // fused global max (saves a 400MB pass)
#pragma unroll
    for (int off = 16; off > 0; off >>= 1)
      mx = fmaxf(mx, __shfl_xor(mx, off, 32));
    if (lane == 0) atomicMax(gmax, ordf(mx));
  }
}

// ---------- row normalize h -> bf16 xn ----------
__global__ void k_rownorm(const float* __restrict__ h, __bf16* __restrict__ xnb, int dim) {
  const int row = blockIdx.x;
  const int t = threadIdx.x;                        // blockDim.x == dim == 256
  float v = h[(long long)row * dim + t];
  float s = v * v;
#pragma unroll
  for (int off = 16; off > 0; off >>= 1) s += __shfl_xor(s, off, 32);
  __shared__ float wsum[8];
  if ((t & 31) == 0) wsum[t >> 5] = s;
  __syncthreads();
  float tot = 0.f;
#pragma unroll
  for (int w = 0; w < 8; ++w) tot += wsum[w];
  const float inv = 1.0f / (sqrtf(tot) + 1e-8f);
  xnb[(long long)row * dim + t] = f2bf(v * inv);
}

// ---------- boost original edges by global max ----------
__global__ void k_boost(float* __restrict__ adj, const int* __restrict__ src,
                        const int* __restrict__ dst, const unsigned* __restrict__ scal,
                        int nE) {
  int e = blockIdx.x * blockDim.x + threadIdx.x;
  if (e >= nE) return;
  const float mx = unordf(scal[0]);
  atomicAdd(adj + (long long)src[e] * NN + dst[e], mx);
}

// ---------- radix-select top-K: 2048-bucket histogram on ordered-uint top bits ----------
__global__ void k_hist(const float4* __restrict__ adj4, long long n4,
                       unsigned* __restrict__ hist) {
  __shared__ unsigned lh[2048];
  for (int i = threadIdx.x; i < 2048; i += blockDim.x) lh[i] = 0;
  __syncthreads();
  const long long stride = (long long)gridDim.x * blockDim.x;
  for (long long i = (long long)blockIdx.x * blockDim.x + threadIdx.x; i < n4; i += stride) {
    float4 v = adj4[i];
    atomicAdd(&lh[ordf(v.x) >> 21], 1u);
    atomicAdd(&lh[ordf(v.y) >> 21], 1u);
    atomicAdd(&lh[ordf(v.z) >> 21], 1u);
    atomicAdd(&lh[ordf(v.w) >> 21], 1u);
  }
  __syncthreads();
  for (int i = threadIdx.x; i < 2048; i += blockDim.x) {
    unsigned cv = lh[i];
    if (cv) atomicAdd(&hist[i], cv);
  }
}

__global__ void k_scan(const unsigned* __restrict__ hist, unsigned* __restrict__ scal, int Kk) {
  if (threadIdx.x | blockIdx.x) return;
  unsigned cum = 0, cgt = 0; int T = 0;
  for (int b = 2047; b >= 0; --b) {
    unsigned h = hist[b];
    if (cum + h >= (unsigned)Kk) { T = b; cgt = cum; break; }
    cum += h;
  }
  scal[1] = (unsigned)T;   // threshold bucket
  scal[2] = cgt;           // count strictly above threshold bucket
  scal[3] = 0;             // gt write counter
  scal[4] = 0;             // eq write counter
}

__global__ void k_select(const float* __restrict__ adj, long long n,
                         unsigned* __restrict__ scal,
                         float* __restrict__ vals, int* __restrict__ vidx, int Kk) {
  const unsigned T = scal[1], cgt = scal[2];
  const long long stride = (long long)gridDim.x * blockDim.x;
  for (long long i = (long long)blockIdx.x * blockDim.x + threadIdx.x; i < n; i += stride) {
    const float v = adj[i];
    const unsigned b = ordf(v) >> 21;
    if (b > T) {
      unsigned p = atomicAdd(&scal[3], 1u);
      vals[p] = v; vidx[p] = (int)i;
    } else if (b == T) {
      unsigned q = atomicAdd(&scal[4], 1u);
      unsigned p = cgt + q;
      if (p < (unsigned)Kk) { vals[p] = v; vidx[p] = (int)i; }
    }
  }
}

// ---------- outputs + segment softmax ----------
__global__ void k_edges(const int* __restrict__ vidx, float* __restrict__ osrc,
                        float* __restrict__ odst, int* __restrict__ segsrc, int Kk) {
  int k = blockIdx.x * blockDim.x + threadIdx.x;
  if (k >= Kk) return;
  const int id = vidx[k];
  const int s = id / NN, d = id % NN;
  osrc[k] = (float)s; odst[k] = (float)d;
  segsrc[k] = s;
}
__global__ void k_segmax(const float* __restrict__ vals, const int* __restrict__ segsrc,
                         unsigned* __restrict__ segmaxb, int Kk) {
  int k = blockIdx.x * blockDim.x + threadIdx.x;
  if (k >= Kk) return;
  atomicMax(&segmaxb[segsrc[k]], ordf(vals[k]));
}
__global__ void k_segexp(const float* __restrict__ vals, const int* __restrict__ segsrc,
                         const unsigned* __restrict__ segmaxb, float* __restrict__ ew,
                         float* __restrict__ segsum, int Kk) {
  int k = blockIdx.x * blockDim.x + threadIdx.x;
  if (k >= Kk) return;
  const int s = segsrc[k];
  const float e = __expf(vals[k] - unordf(segmaxb[s]));
  ew[k] = e;
  atomicAdd(&segsum[s], e);
}
__global__ void k_segnorm(float* __restrict__ ew, const int* __restrict__ segsrc,
                          const float* __restrict__ segsum, int Kk) {
  int k = blockIdx.x * blockDim.x + threadIdx.x;
  if (k >= Kk) return;
  ew[k] = ew[k] / segsum[segsrc[k]];
}

// ---------- orchestration ----------
extern "C" void kernel_launch(void* const* d_in, const int* in_sizes, int n_in,
                              void* d_out, int out_size, void* d_ws, size_t ws_size,
                              hipStream_t stream) {
  const float* x    = (const float*)d_in[0];
  const int*   ei   = (const int*)d_in[1];
  const int*   esrc = ei;
  const int*   edst = ei + NEDGE;
  const float* W1a[3] = {(const float*)d_in[2],  (const float*)d_in[6],  (const float*)d_in[10]};
  const float* b1a[3] = {(const float*)d_in[3],  (const float*)d_in[7],  (const float*)d_in[11]};
  const float* W2a[3] = {(const float*)d_in[4],  (const float*)d_in[8],  (const float*)d_in[12]};
  const float* b2a[3] = {(const float*)d_in[5],  (const float*)d_in[9],  (const float*)d_in[13]};

  char* ws = (char*)d_ws;
  size_t off = 0;
  auto alloc = [&](size_t bytes) -> void* {
    void* p = ws + off;
    off += (bytes + 255) & ~(size_t)255;
    return p;
  };
  float*    adj     = (float*)alloc((size_t)NN * NN * 4);       // 400 MB
  float*    hA      = (float*)alloc((size_t)NN * HIDD * 4);
  float*    hB      = (float*)alloc((size_t)NN * HIDD * 4);
  float*    agg     = (float*)alloc((size_t)NN * HIDD * 4);
  __bf16*   zbf     = (__bf16*)alloc((size_t)NN * HIDD * 2);
  __bf16*   y1bf    = (__bf16*)alloc((size_t)NN * HIDD * 2);
  __bf16*   xnbf    = (__bf16*)alloc((size_t)NN * HIDD * 2);
  __bf16*   W1t     = (__bf16*)alloc((size_t)HIDD * HIDD * 2);
  __bf16*   W2t     = (__bf16*)alloc((size_t)HIDD * HIDD * 2);
  float*    vals    = (float*)alloc((size_t)KSEL * 4);
  int*      vidx    = (int*)alloc((size_t)KSEL * 4);
  int*      segsrc  = (int*)alloc((size_t)KSEL * 4);
  unsigned* segmaxb = (unsigned*)alloc((size_t)NN * 4);
  float*    segsum  = (float*)alloc((size_t)NN * 4);
  unsigned* hist    = (unsigned*)alloc(2048 * 4);
  unsigned* scal    = (unsigned*)alloc(256);

  auto blocks = [](long long n) { return (unsigned)((n + 255) / 256); };
  const dim3 gRows((NN + 127) / 128, 1);             // 8 waves/block stacked in M

  // init (deterministic each call; ws is not re-poisoned between replays)
  k_fill_u32<<<1, 64, 0, stream>>>(scal, 0u, 64);
  k_fill_u32<<<8, 256, 0, stream>>>(hist, 0u, 2048);
  k_fill_u32<<<blocks(NN), 256, 0, stream>>>(segmaxb, 0u, NN);
  k_fill_u32<<<blocks(NN), 256, 0, stream>>>((unsigned*)segsum, 0u, NN);

  // ---- GIN encoder: 3 layers ----
  const float* hcur = x;
  float* ping[2] = {hA, hB};
  for (int l = 0; l < 3; ++l) {
    const int din = (l == 0) ? INDIM : HIDD;
    k_fill_u32<<<blocks((long long)NN * din), 256, 0, stream>>>((unsigned*)agg, 0u, (long long)NN * din);
    k_agg<<<blocks((long long)NEDGE * (din >> 2)), 256, 0, stream>>>(hcur, esrc, edst, agg, din, NEDGE);
    k_add_cvt<<<blocks((long long)NN * din), 256, 0, stream>>>(hcur, agg, zbf, (long long)NN * din);
    k_cvt_wt<<<blocks((long long)din * HIDD), 256, 0, stream>>>(W1a[l], W1t, din, HIDD);

    dim3 g1((NN + 127) / 128, HIDD / 64);
    if (din == INDIM) {
      k_gemm_t<INDIM, 4, true, true, false, true, false><<<g1, 256, 0, stream>>>(
          zbf, W1t, b1a[l], nullptr, y1bf, nullptr, NN, HIDD, 0);
    } else {
      k_gemm_t<HIDD, 4, true, true, false, true, false><<<g1, 256, 0, stream>>>(
          zbf, W1t, b1a[l], nullptr, y1bf, nullptr, NN, HIDD, 0);
    }
    k_cvt_wt<<<blocks((long long)HIDD * HIDD), 256, 0, stream>>>(W2a[l], W2t, HIDD, HIDD);
    float* hnext = ping[l & 1];
    k_gemm_t<HIDD, 4, true, true, true, false, false><<<g1, 256, 0, stream>>>(
        y1bf, W2t, b2a[l], hnext, nullptr, nullptr, NN, HIDD, 0);
    hcur = hnext;
  }

  // ---- cosine similarity Gram matrix (WMMA), with fused global-max ----
  k_rownorm<<<NN, 256, 0, stream>>>(hcur, xnbf, HIDD);
  {
    const int nFull = NN / 64;                       // 156 full 64-wide strips
    dim3 gsMain((NN + 127) / 128, nFull);
    k_gemm_t<HIDD, 4, false, false, true, false, true><<<gsMain, 256, 0, stream>>>(
        xnbf, xnbf, nullptr, adj, nullptr, &scal[0], NN, NN, 0);
    dim3 gsTail((NN + 127) / 128, 1);                // remaining 16 columns at 9984
    k_gemm_t<HIDD, 1, false, false, true, false, true><<<gsTail, 256, 0, stream>>>(
        xnbf, xnbf, nullptr, adj, nullptr, &scal[0], NN, NN, nFull * 64);
  }

  // ---- boost original edges by adj.max() ----
  k_boost<<<blocks(NEDGE), 256, 0, stream>>>(adj, esrc, edst, scal, NEDGE);

  // ---- exact top-K radix select ----
  k_hist<<<2048, 256, 0, stream>>>((const float4*)adj, (long long)NN * NN / 4, hist);
  k_scan<<<1, 1, 0, stream>>>(hist, scal, KSEL);
  k_select<<<4096, 256, 0, stream>>>(adj, (long long)NN * NN, scal, vals, vidx, KSEL);

  // ---- outputs: [x | new_src | new_dst | edge_weight] ----
  float* out  = (float*)d_out;
  float* osrc = out + (size_t)NN * INDIM;
  float* odst = osrc + KSEL;
  float* oew  = odst + KSEL;
  k_copy_f32<<<blocks((long long)NN * INDIM), 256, 0, stream>>>(x, out, (long long)NN * INDIM);
  k_edges<<<blocks(KSEL), 256, 0, stream>>>(vidx, osrc, odst, segsrc, KSEL);

  // ---- segment softmax grouped by new_src ----
  k_segmax<<<blocks(KSEL), 256, 0, stream>>>(vals, segsrc, segmaxb, KSEL);
  k_segexp<<<blocks(KSEL), 256, 0, stream>>>(vals, segsrc, segmaxb, oew, segsum, KSEL);
  k_segnorm<<<blocks(KSEL), 256, 0, stream>>>(oew, segsrc, segsum, KSEL);
}
